// LSTMModel_80779744903374
// MI455X (gfx1250) — matrix operand order repaired and checked by reading, not compiled, not run on hardware
//
#include <hip/hip_runtime.h>
#include <hip/hip_bf16.h>
#include <stdint.h>

// ---------------------------------------------------------------------------
// LSTM sentiment model on MI455X (gfx1250).
//  B=256, T=512, E=128, H=256, 4H=1024.
//  Strategy: fused [h|x_t] @ [W_hh|W_ih]^T recurrence (K=384) in bf16 WMMA,
//  persistent 16-WG kernel, TDM tensor loads for per-step activations,
//  cell state resident in VGPRs, atomic grid barrier between steps.
// ---------------------------------------------------------------------------

typedef __bf16 bf16_t;
typedef bf16_t v16bf __attribute__((ext_vector_type(16)));
typedef bf16_t v8bf  __attribute__((ext_vector_type(8)));
typedef float  v8f   __attribute__((ext_vector_type(8)));
typedef unsigned int u32x4 __attribute__((ext_vector_type(4)));
typedef int          i32x8 __attribute__((ext_vector_type(8)));
typedef int          i32x4 __attribute__((ext_vector_type(4)));

#define B_SZ   256
#define T_SZ   512
#define E_SZ   128
#define H_SZ   256
#define G4     1024          // 4*H
#define KTOT   384           // H + E  (concat K depth)
#define NWG    16            // persistent workgroups (one per WGP)
#define WAVES  16            // waves per WG (one batch tile each)

// workspace layout (bytes)
#define OFF_CNT   0u
#define OFF_BSUM  256u                            // 1024 f32   -> 4 KB
#define OFF_H     8192u                           // 256x256 bf16 -> 128 KB
#define OFF_W     139264u                         // 1024x384 bf16 -> 768 KB
#define OFF_E     1048576u                        // 256x512x128 bf16 -> 33.5 MB
// total ~34.6 MB

// --------------------------- TDM descriptor helpers -------------------------
static __device__ inline u32x4 tdm_g0(unsigned lds_addr, unsigned long long gaddr) {
  u32x4 g;
  g.x = 1u;                                        // count=1, user-mode load
  g.y = lds_addr;                                  // LDS byte address
  g.z = (unsigned)(gaddr & 0xFFFFFFFFull);         // global addr [31:0]
  g.w = (unsigned)((gaddr >> 32) & 0x1FFFFFFull)   // global addr [56:32]
      | (2u << 30);                                // type = 2 ("image")
  return g;
}

static __device__ inline i32x8 tdm_g1(unsigned pad_en, unsigned pad_interval,
                                      unsigned pad_amount,
                                      unsigned long long dim0,
                                      unsigned long long dim1,
                                      unsigned tile0, unsigned tile1,
                                      unsigned long long stride0) {
  // data_size = 1 (2-byte elements); no multicast; no iterate.
  unsigned w0 = (1u << 16) | (pad_en << 20) | (pad_interval << 22) | (pad_amount << 25);
  unsigned w1 = (unsigned)((dim0 & 0xFFFFull) << 16);          // tensor_dim0[15:0]
  unsigned w2 = (unsigned)((dim0 >> 16) & 0xFFFFull)           // tensor_dim0[31:16]
              | (unsigned)((dim1 & 0xFFFFull) << 16);          // tensor_dim1[15:0]
  unsigned w3 = (unsigned)((dim1 >> 16) & 0xFFFFull)           // tensor_dim1[31:16]
              | (tile0 << 16);                                 // tile_dim0
  unsigned w4 = (tile1 & 0xFFFFu);                             // tile_dim1 (tile_dim2=0)
  unsigned w5 = (unsigned)(stride0 & 0xFFFFFFFFull);           // dim0_stride[31:0]
  unsigned w6 = (unsigned)((stride0 >> 32) & 0xFFFFull);       // dim0_stride[47:32]
  unsigned w7 = 0u;                                            // dim1_stride unused (2D)
  i32x8 g;
  g[0] = (int)w0; g[1] = (int)w1; g[2] = (int)w2; g[3] = (int)w3;
  g[4] = (int)w4; g[5] = (int)w5; g[6] = (int)w6; g[7] = (int)w7;
  return g;
}

// ------------------------------- math helpers -------------------------------
static __device__ inline float sigmoid_f(float x) {
  return 1.0f / (1.0f + __expf(-x));
}
static __device__ inline float tanh_f(float x) {
  float e = __expf(2.0f * x);
  return (e - 1.0f) / (e + 1.0f);
}

// ------------------------------- prep kernels -------------------------------
__global__ void prep_misc(const float* __restrict__ b_ih, const float* __restrict__ b_hh,
                          float* __restrict__ bsum, bf16_t* __restrict__ hglob,
                          unsigned* __restrict__ counter) {
  int tid = threadIdx.x;
  if (tid == 0) *counter = 0u;
  for (int i = tid; i < G4; i += blockDim.x) bsum[i] = b_ih[i] + b_hh[i];
  for (int i = tid; i < B_SZ * H_SZ; i += blockDim.x) hglob[i] = (bf16_t)0.0f;
}

// Wcat[g][0:256) = W_hh[g][:], Wcat[g][256:384) = W_ih[g][:], bf16
__global__ void prep_wcat(const float* __restrict__ W_hh, const float* __restrict__ W_ih,
                          bf16_t* __restrict__ wcat) {
  int g = blockIdx.x;          // 0..1023
  int c = threadIdx.x;         // 0..383
  float v = (c < H_SZ) ? W_hh[g * H_SZ + c] : W_ih[g * E_SZ + (c - H_SZ)];
  wcat[g * KTOT + c] = (bf16_t)v;
}

// embedded[b*T+t][e] = emb_table[x[b,t]][e] as bf16; layout [B][T*E]
__global__ void prep_emb(const int* __restrict__ x, const float* __restrict__ etab,
                         bf16_t* __restrict__ embg) {
  long long bt = blockIdx.x;                 // 0..131071
  int e = threadIdx.x;                       // 0..127
  long long tok = (long long)x[bt];
  embg[bt * E_SZ + e] = (bf16_t)etab[tok * E_SZ + e];
}

// --------------------------- persistent LSTM kernel -------------------------
__global__ __launch_bounds__(512, 1)
void lstm_persistent(const bf16_t* __restrict__ wcat,   // [1024][384]
                     const float*  __restrict__ bsum,   // [1024]
                     const bf16_t* __restrict__ embg,   // [B][T*E]
                     bf16_t* __restrict__ hglob,        // [256][256]
                     unsigned* __restrict__ counter) {
  extern __shared__ char smem[];
  bf16_t* shA = (bf16_t*)smem;                         // [256][384] h|emb concat
  bf16_t* shW = (bf16_t*)(smem + B_SZ * KTOT * 2);     // [64][384] this WG's W rows

  const int nj   = blockIdx.x;           // hidden-j tile 0..15
  const int wave = threadIdx.x >> 5;     // batch tile m 0..15
  const int lane = threadIdx.x & 31;
  const int hi8  = (lane >> 4) << 3;     // 0 or 8 (C/D row offset for lanes 16..31)
  const int ncol = lane & 15;            // N within tile
  const int tid  = threadIdx.x;

  // Stage this WG's 64 weight rows (4 gates x 16 j's) into LDS once.
  for (int idx = tid; idx < 64 * KTOT; idx += blockDim.x) {
    int r = idx / KTOT, c = idx - r * KTOT;
    int g = ((r >> 4) << 8) + (nj << 4) + (r & 15);    // gate*256 + j
    shW[idx] = wcat[g * KTOT + c];
  }
  __syncthreads();

  // Biases (b_ih + b_hh) for this wave's 4 gate tiles.
  const float bI = bsum[0 * H_SZ + (nj << 4) + ncol];
  const float bF = bsum[1 * H_SZ + (nj << 4) + ncol];
  const float bG = bsum[2 * H_SZ + (nj << 4) + ncol];
  const float bO = bsum[3 * H_SZ + (nj << 4) + ncol];

  v8f c = {};                                          // cell state, VGPR-resident
  unsigned phase = 0;

  const unsigned ldsA = (unsigned)(size_t)shA;
  const unsigned long long hga = (unsigned long long)(size_t)hglob;
  const unsigned long long ega = (unsigned long long)(size_t)embg;
  const i32x4 z4 = {0, 0, 0, 0};
  const i32x8 z8 = {0, 0, 0, 0, 0, 0, 0, 0};

  const bf16_t* arow = shA + (((wave << 4) + ncol) * KTOT);

  for (int t = 0; t < T_SZ; ++t) {
    // -- stage activations: h (128KB) + emb[:,t,:] (64KB) via TDM ------------
    if (wave == 0) {
#if __has_builtin(__builtin_amdgcn_tensor_load_to_lds)
      // h: 256x256 tile, LDS rows padded by 128 elems (64 DW) to stride 384.
      __builtin_amdgcn_tensor_load_to_lds(
          tdm_g0(ldsA, hga),
          tdm_g1(/*pad*/1, /*interval 128DW*/6, /*amount 64DW*/63,
                 /*dim0*/H_SZ, /*dim1*/B_SZ, /*tile0*/H_SZ, /*tile1*/B_SZ,
                 /*stride0*/H_SZ),
          z4, z4, z8, 0);
      // emb column t: 256 rows of 128 elems, row stride T*E, LDS rows padded
      // by 256 elems (128 DW) and interleaved at offset 512B after each h row.
      __builtin_amdgcn_tensor_load_to_lds(
          tdm_g0(ldsA + (unsigned)(H_SZ * 2), ega + (unsigned long long)t * (E_SZ * 2)),
          tdm_g1(/*pad*/1, /*interval 64DW*/5, /*amount 128DW*/127,
                 /*dim0*/(unsigned long long)T_SZ * E_SZ, /*dim1*/B_SZ,
                 /*tile0*/E_SZ, /*tile1*/B_SZ,
                 /*stride0*/(unsigned long long)T_SZ * E_SZ),
          z4, z4, z8, 0);
      __builtin_amdgcn_s_wait_tensorcnt(0);
#else
      // fallback: cooperative wave copy (compiles everywhere)
      for (int i = lane; i < B_SZ * H_SZ / 2; i += 32) {
        int r = i / (H_SZ / 2), cc = (i - r * (H_SZ / 2)) * 2;
        *(uint32_t*)&shA[r * KTOT + cc] = *(const uint32_t*)&hglob[r * H_SZ + cc];
      }
      for (int i = lane; i < B_SZ * E_SZ / 2; i += 32) {
        int r = i / (E_SZ / 2), cc = (i - r * (E_SZ / 2)) * 2;
        *(uint32_t*)&shA[r * KTOT + H_SZ + cc] =
            *(const uint32_t*)&embg[(long long)r * T_SZ * E_SZ + (long long)t * E_SZ + cc];
      }
#endif
    }
    __syncthreads();   // LDS activation tile ready for all 16 waves

    // -- gates = [h|x] @ Wcat^T + bias, four 16x16 tiles, K=384 --------------
    v8f aI = {bI, bI, bI, bI, bI, bI, bI, bI};
    v8f aF = {bF, bF, bF, bF, bF, bF, bF, bF};
    v8f aG = {bG, bG, bG, bG, bG, bG, bG, bG};
    v8f aO = {bO, bO, bO, bO, bO, bO, bO, bO};

#pragma unroll 4
    for (int kc = 0; kc < KTOT / 32; ++kc) {
      // A fragment (16x32 bf16): lane holds row M=ncol, 16 contiguous K
      // starting at kb = kc*32 + (lane<16 ? 0 : 16)  (matches B).
      int kb = kc * 32 + ((lane >> 4) << 4);
      v16bf a = *(const v16bf*)(arow + kb);
      // B fragment (32x16 bf16): lane holds column N=ncol, 16 contiguous K
      // starting at kc*32 + (lane<16 ? 0 : 16)  == a row slice of Wcat.
      int kw = kc * 32 + ((lane >> 4) << 4);
      v16bf wI = *(const v16bf*)(shW + (0 * 16 + ncol) * KTOT + kw);
      v16bf wF = *(const v16bf*)(shW + (1 * 16 + ncol) * KTOT + kw);
      v16bf wG = *(const v16bf*)(shW + (2 * 16 + ncol) * KTOT + kw);
      v16bf wO = *(const v16bf*)(shW + (3 * 16 + ncol) * KTOT + kw);

      aI = __builtin_amdgcn_wmma_f32_16x16x32_bf16(false, a, false, wI, (short)0, aI, false, false);
      aF = __builtin_amdgcn_wmma_f32_16x16x32_bf16(false, a, false, wF, (short)0, aF, false, false);
      aG = __builtin_amdgcn_wmma_f32_16x16x32_bf16(false, a, false, wG, (short)0, aG, false, false);
      aO = __builtin_amdgcn_wmma_f32_16x16x32_bf16(false, a, false, wO, (short)0, aO, false, false);
    }

    // -- elementwise LSTM cell update (c stays in VGPRs) ---------------------
    v8f h;
#pragma unroll
    for (int r = 0; r < 8; ++r) {
      float iv = sigmoid_f(aI[r]);
      float fv = sigmoid_f(aF[r]);
      float gv = tanh_f(aG[r]);
      float ov = sigmoid_f(aO[r]);
      float cv = fv * c[r] + iv * gv;
      c[r] = cv;
      h[r] = ov * tanh_f(cv);
    }

    // grid barrier #1: all WGs finished reading hglob for step t
    __syncthreads();
    if (tid == 0) {
      __hip_atomic_fetch_add(counter, 1u, __ATOMIC_ACQ_REL, __HIP_MEMORY_SCOPE_AGENT);
      unsigned tgt = (unsigned)NWG * (++phase);
      while (__hip_atomic_load(counter, __ATOMIC_ACQUIRE, __HIP_MEMORY_SCOPE_AGENT) < tgt)
        __builtin_amdgcn_s_sleep(2);
    }
    __syncthreads();

    // -- publish h(t): D layout -> hglob[b][j] bf16 --------------------------
    {
      int bbase = (wave << 4) + hi8;
      int col = (nj << 4) + ncol;
#pragma unroll
      for (int r = 0; r < 8; ++r)
        hglob[(bbase + r) * H_SZ + col] = (bf16_t)h[r];
    }
    asm volatile("s_wait_storecnt 0x0" ::: "memory");
    __threadfence();

    // grid barrier #2: h(t) visible device-wide before next step's TDM
    __syncthreads();
    if (tid == 0) {
      __hip_atomic_fetch_add(counter, 1u, __ATOMIC_ACQ_REL, __HIP_MEMORY_SCOPE_AGENT);
      unsigned tgt = (unsigned)NWG * (++phase);
      while (__hip_atomic_load(counter, __ATOMIC_ACQUIRE, __HIP_MEMORY_SCOPE_AGENT) < tgt)
        __builtin_amdgcn_s_sleep(2);
    }
    __syncthreads();
  }
}

// ------------------------------- output head --------------------------------
__global__ void fc_kernel(const bf16_t* __restrict__ hglob, const float* __restrict__ wfc,
                          const float* __restrict__ bfc, float* __restrict__ out) {
  int b = threadIdx.x;         // 0..255
  float s = bfc[0];
  for (int j = 0; j < H_SZ; ++j) s += (float)hglob[b * H_SZ + j] * wfc[j];
  out[b] = sigmoid_f(s);
}

// ------------------------------- launcher -----------------------------------
extern "C" void kernel_launch(void* const* d_in, const int* in_sizes, int n_in,
                              void* d_out, int out_size, void* d_ws, size_t ws_size,
                              hipStream_t stream) {
  const int*   x    = (const int*)d_in[0];
  const float* etab = (const float*)d_in[1];
  const float* W_ih = (const float*)d_in[2];
  const float* W_hh = (const float*)d_in[3];
  const float* b_ih = (const float*)d_in[4];
  const float* b_hh = (const float*)d_in[5];
  const float* W_fc = (const float*)d_in[6];
  const float* b_fc = (const float*)d_in[7];
  float* out = (float*)d_out;

  char* ws = (char*)d_ws;
  unsigned* counter = (unsigned*)(ws + OFF_CNT);
  float*    bsum    = (float*)(ws + OFF_BSUM);
  bf16_t*   hglob   = (bf16_t*)(ws + OFF_H);
  bf16_t*   wcat    = (bf16_t*)(ws + OFF_W);
  bf16_t*   embg    = (bf16_t*)(ws + OFF_E);

  prep_misc<<<1, 1024, 0, stream>>>(b_ih, b_hh, bsum, hglob, counter);
  prep_wcat<<<G4, KTOT, 0, stream>>>(W_hh, W_ih, wcat);
  prep_emb<<<B_SZ * T_SZ, E_SZ, 0, stream>>>(x, etab, embg);

  const size_t lds_bytes = (size_t)B_SZ * KTOT * 2 + (size_t)64 * KTOT * 2; // 240 KB
  lstm_persistent<<<NWG, 32 * WAVES, lds_bytes, stream>>>(wcat, bsum, embg, hglob, counter);

  fc_kernel<<<1, B_SZ, 0, stream>>>(hglob, W_fc, b_fc, out);
}
